// ThinkerDecoderLayer_11184094838835
// MI455X (gfx1250) — matrix operand order repaired and checked by reading, not compiled, not run on hardware
//
#include <hip/hip_runtime.h>
#include <hip/hip_bf16.h>
#include <math.h>

// ------------------------- problem constants -------------------------
constexpr int H    = 2048;
constexpr int NH   = 16;
constexpr int HD   = H / NH;     // 128
constexpr int I_FF = 8192;
constexpr int B_   = 2;
constexpr int T_   = 2048;
constexpr int NTOK = B_ * T_;    // 4096
__constant__ float kEps = 1e-6f;

// ------------------------- CDNA5 WMMA types --------------------------
typedef __bf16 v16bf __attribute__((ext_vector_type(16)));
typedef float  v8f   __attribute__((ext_vector_type(8)));
typedef unsigned int v4u __attribute__((ext_vector_type(4)));
typedef int gv4i __attribute__((vector_size(16)));   // matches async-LDS builtin param

union Frag {              // 32 bytes: one bf16 A/B fragment for 16x16x32
    v4u  u4[2];
    v16bf bf;
};

__device__ __forceinline__ v8f wmma_bf16(v16bf a, v16bf b, v8f c) {
    // D(f32 16x16) = A(bf16 16x32) * B(bf16 32x16) + C
    return __builtin_amdgcn_wmma_f32_16x16x32_bf16(
        /*neg_a=*/false, a, /*neg_b=*/false, b,
        /*c_mod=*/(short)0, c, /*reuse_a=*/false, /*reuse_b=*/false);
}

__device__ __forceinline__ unsigned short f32_to_bf16(float f) {
    unsigned int u = __float_as_uint(f);
    unsigned int r = 0x7FFFu + ((u >> 16) & 1u);   // round-to-nearest-even
    u += r;
    return (unsigned short)(u >> 16);
}

__device__ __forceinline__ float rowmax16(float v) {
    for (int m = 1; m < 16; m <<= 1) v = fmaxf(v, __shfl_xor(v, m, 32));
    return v;
}
__device__ __forceinline__ float rowsum16(float v) {
    for (int m = 1; m < 16; m <<= 1) v += __shfl_xor(v, m, 32);
    return v;
}

// ---------------- CDNA5 async global->LDS copy (guarded) -------------
#if defined(__has_builtin)
#if __has_builtin(__builtin_amdgcn_global_load_async_to_lds_b128)
#define USE_ASYNC_LDS 1
#endif
#endif

__device__ __forceinline__ void cp16_to_lds(unsigned short* l, const unsigned short* g) {
#ifdef USE_ASYNC_LDS
    __builtin_amdgcn_global_load_async_to_lds_b128(
        (__attribute__((address_space(1))) gv4i*)(g),
        (__attribute__((address_space(3))) gv4i*)(l), 0, 0);
#else
    *(v4u*)l = *(const v4u*)g;
#endif
}

__device__ __forceinline__ void wait_async_copies() {
#ifdef USE_ASYNC_LDS
#if __has_builtin(__builtin_amdgcn_s_wait_asynccnt)
    __builtin_amdgcn_s_wait_asynccnt(0);
#else
    asm volatile("s_wait_asynccnt 0x0" ::: "memory");
#endif
#endif
}

// ------------------------- elementwise kernels -----------------------
__global__ void f32_to_bf16_kernel(const float* __restrict__ in,
                                   unsigned short* __restrict__ out, size_t n) {
    for (size_t i = (size_t)blockIdx.x * blockDim.x + threadIdx.x; i < n;
         i += (size_t)gridDim.x * blockDim.x)
        out[i] = f32_to_bf16(in[i]);
}

__global__ void rmsnorm_bf16_kernel(const float* __restrict__ x,
                                    const float* __restrict__ g,
                                    unsigned short* __restrict__ out, int ncols) {
    __shared__ float red[256];
    int row = blockIdx.x;
    const float* xr = x + (size_t)row * ncols;
    float ss = 0.f;
    for (int c = threadIdx.x; c < ncols; c += blockDim.x) {
        float v = xr[c];
        ss += v * v;
    }
    red[threadIdx.x] = ss;
    __syncthreads();
    for (int s = 128; s > 0; s >>= 1) {
        if ((int)threadIdx.x < s) red[threadIdx.x] += red[threadIdx.x + s];
        __syncthreads();
    }
    float inv = rsqrtf(red[0] / (float)ncols + kEps);
    for (int c = threadIdx.x; c < ncols; c += blockDim.x)
        out[(size_t)row * ncols + c] = f32_to_bf16(xr[c] * inv * g[c]);
}

// interleaved RoPE on q and k; position == token index (b*T + t)
__global__ void rope_kernel(const float* __restrict__ qf, const float* __restrict__ kf,
                            const float* __restrict__ cosc, const float* __restrict__ sinc,
                            unsigned short* __restrict__ qb, unsigned short* __restrict__ kb) {
    const size_t npairs = (size_t)NTOK * (H / 2);
    for (size_t i = (size_t)blockIdx.x * blockDim.x + threadIdx.x; i < npairs;
         i += (size_t)gridDim.x * blockDim.x) {
        size_t tok = i / (H / 2);
        int cpair  = (int)(i % (H / 2));
        int c0     = cpair * 2;
        int d      = c0 % HD;                 // position within a head
        size_t pos = tok;                     // arange(B*T)
        float c_0 = cosc[pos * HD + d];
        float c_1 = cosc[pos * HD + d + 1];
        float s_0 = sinc[pos * HD + d];
        float s_1 = sinc[pos * HD + d + 1];
        size_t base = tok * H + c0;
        float q0 = qf[base], q1 = qf[base + 1];
        qb[base]     = f32_to_bf16(q0 * c_0 - q1 * s_0);
        qb[base + 1] = f32_to_bf16(q1 * c_1 + q0 * s_1);
        float k0 = kf[base], k1 = kf[base + 1];
        kb[base]     = f32_to_bf16(k0 * c_0 - k1 * s_0);
        kb[base + 1] = f32_to_bf16(k1 * c_1 + k0 * s_1);
    }
}

// V [tok][H] f32  ->  Vt [b][h][d][t] bf16  (keys contiguous for PV B-frags)
__global__ void vtrans_kernel(const float* __restrict__ vf,
                              unsigned short* __restrict__ vt) {
    const size_t n = (size_t)NTOK * H;
    for (size_t i = (size_t)blockIdx.x * blockDim.x + threadIdx.x; i < n;
         i += (size_t)gridDim.x * blockDim.x) {
        size_t tok = i / H;
        int c  = (int)(i % H);
        int hh = c / HD, d = c % HD;
        int b  = (int)(tok / T_), t = (int)(tok % T_);
        vt[(((size_t)b * NH + hh) * HD + d) * T_ + t] = f32_to_bf16(vf[i]);
    }
}

// ------------------------- WMMA GEMM ---------------------------------
// C[M,N] = A[M,K](bf16) * W[N,K](bf16)^T, f32 accumulate.
// Block tile 64(M) x 256(N) x 32(K); 8 waves as 2(M) x 4(N); each wave
// owns a 32x64 register tile = 2x4 WMMA accumulators. A/B K-slices are
// double-buffered in LDS, staged with async global->LDS when available.
// mode 0: C=f32  1: C=f32+resid  2: C=bf16 silu(acc)
#define GEMM_BM 64
#define GEMM_BN 256
#define GEMM_BK 32

__global__ __launch_bounds__(256)
void gemm_bf16_kernel(const unsigned short* __restrict__ A,
                      const unsigned short* __restrict__ Wt,
                      float* __restrict__ Cf,
                      unsigned short* __restrict__ Cb,
                      const float* __restrict__ resid,
                      int M, int N, int K, int mode) {
    __shared__ unsigned short ldsA[2][GEMM_BM * GEMM_BK];   // 2 x 4 KB
    __shared__ unsigned short ldsB[2][GEMM_BN * GEMM_BK];   // 2 x 16 KB

    int tid      = threadIdx.x;
    int w        = tid >> 5;
    int lane     = tid & 31;
    int laneM    = lane & 15;
    int laneHalf = lane >> 4;
    int wm = w >> 2;            // 0..1 (M direction)
    int wn = w & 3;             // 0..3 (N direction)
    int mblk = blockIdx.x * GEMM_BM;
    int nblk = blockIdx.y * GEMM_BN;

    // staging: each thread copies 1 A chunk + 4 B chunks of 16B per K-step
    int srow   = tid >> 2;              // 0..63
    int schunk = (tid & 3) * 8;         // ushort offset within a 32-wide row
    const unsigned short* gA = A  + (size_t)(mblk + srow) * K + schunk;
    const unsigned short* gB = Wt + (size_t)(nblk + srow) * K + schunk;

    auto stage = [&](int buf, int k0) {
        cp16_to_lds(&ldsA[buf][srow * GEMM_BK + schunk], gA + k0);
        for (int j = 0; j < 4; ++j)
            cp16_to_lds(&ldsB[buf][(srow + 64 * j) * GEMM_BK + schunk],
                        gB + (size_t)(64 * j) * K + k0);
    };

    v8f acc[2][4];
    v8f z = {};
    for (int mt = 0; mt < 2; ++mt)
        for (int nt = 0; nt < 4; ++nt) acc[mt][nt] = z;

    stage(0, 0);
    int buf = 0;
    for (int k0 = 0; k0 < K; k0 += GEMM_BK) {
        wait_async_copies();
        __syncthreads();                      // buffer `buf` is ready
        if (k0 + GEMM_BK < K) stage(buf ^ 1, k0 + GEMM_BK);

        Frag afr[2], bfr[4];
        for (int mt = 0; mt < 2; ++mt) {
            const unsigned short* p =
                &ldsA[buf][(wm * 32 + mt * 16 + laneM) * GEMM_BK + laneHalf * 8];
            afr[mt].u4[0] = *(const v4u*)p;
            afr[mt].u4[1] = *(const v4u*)(p + 16);
        }
        for (int nt = 0; nt < 4; ++nt) {
            const unsigned short* p =
                &ldsB[buf][(wn * 64 + nt * 16 + laneM) * GEMM_BK + laneHalf * 8];
            bfr[nt].u4[0] = *(const v4u*)p;
            bfr[nt].u4[1] = *(const v4u*)(p + 16);
        }
        for (int mt = 0; mt < 2; ++mt)
            for (int nt = 0; nt < 4; ++nt)
                acc[mt][nt] = wmma_bf16(afr[mt].bf, bfr[nt].bf, acc[mt][nt]);
        buf ^= 1;
    }

    for (int mt = 0; mt < 2; ++mt)
        for (int nt = 0; nt < 4; ++nt)
            for (int j = 0; j < 8; ++j) {
                int m = mblk + wm * 32 + mt * 16 + j + laneHalf * 8;
                int n = nblk + wn * 64 + nt * 16 + laneM;
                size_t idx = (size_t)m * N + n;
                float v = acc[mt][nt][j];
                if (mode == 1) {
                    Cf[idx] = v + resid[idx];
                } else if (mode == 2) {
                    float s = v / (1.f + __expf(-v));            // silu
                    Cb[idx] = f32_to_bf16(s);
                } else {
                    Cf[idx] = v;
                }
            }
}

// ------------------------- flash attention ---------------------------
// grid = (B*NH, T/128); block = 256 = 8 waves; each wave owns 16 q rows.
__global__ void attn_kernel(const unsigned short* __restrict__ qb,
                            const unsigned short* __restrict__ kb,
                            const unsigned short* __restrict__ vt,
                            unsigned short* __restrict__ ob) {
    __shared__ unsigned short plds[8][16 * 32];   // per-wave P staging (bf16)
    int bh = blockIdx.x;
    int b  = bh / NH, h = bh % NH;
    int w        = threadIdx.x >> 5;
    int lane     = threadIdx.x & 31;
    int laneM    = lane & 15;
    int laneHalf = lane >> 4;
    int qt0 = (blockIdx.y * 8 + w) * 16;
    const float scale = 0.08838834764831845f;     // 1/sqrt(128)

    // Q tile: 4 A-fragments covering d = 0..127
    Frag qa[4];
    {
        int tq = qt0 + laneM;
        const unsigned short* qp = qb + ((size_t)(b * T_ + tq) * H + h * HD);
        for (int f = 0; f < 4; ++f) {
            int d0 = f * 32 + laneHalf * 8;
            qa[f].u4[0] = *(const v4u*)(qp + d0);
            qa[f].u4[1] = *(const v4u*)(qp + d0 + 16);
        }
    }

    v8f o[8];
    v8f z = {};
    for (int dt = 0; dt < 8; ++dt) o[dt] = z;
    float mrun[8], lsum[8];
    for (int j = 0; j < 8; ++j) { mrun[j] = -3.0e38f; lsum[j] = 0.f; }

    for (int kc = 0; kc < qt0 + 16; kc += 32) {     // causal bound
        // ---- scores: S[16][32] = Q * K^T, two 16x16 C tiles
        v8f s0 = z, s1 = z;
        for (int f = 0; f < 4; ++f) {
            int d0  = f * 32 + laneHalf * 8;
            int tk0 = kc + laneM;
            int tk1 = kc + 16 + laneM;
            const unsigned short* kp0 = kb + ((size_t)(b * T_ + tk0) * H + h * HD) + d0;
            const unsigned short* kp1 = kb + ((size_t)(b * T_ + tk1) * H + h * HD) + d0;
            Frag k0f, k1f;
            k0f.u4[0] = *(const v4u*)kp0;  k0f.u4[1] = *(const v4u*)(kp0 + 16);
            k1f.u4[0] = *(const v4u*)kp1;  k1f.u4[1] = *(const v4u*)(kp1 + 16);
            s0 = wmma_bf16(qa[f].bf, k0f.bf, s0);
            s1 = wmma_bf16(qa[f].bf, k1f.bf, s1);
        }
        // ---- online softmax over the 32-key chunk
        for (int j = 0; j < 8; ++j) {
            int q = qt0 + j + laneHalf * 8;
            float a0 = s0[j] * scale + ((kc + laneM)      > q ? -1.0e9f : 0.f);
            float a1 = s1[j] * scale + ((kc + 16 + laneM) > q ? -1.0e9f : 0.f);
            float cm   = rowmax16(fmaxf(a0, a1));
            float mnew = fmaxf(mrun[j], cm);
            float corr = __expf(mrun[j] - mnew);
            mrun[j] = mnew;
            float p0 = __expf(a0 - mnew);
            float p1 = __expf(a1 - mnew);
            lsum[j] = lsum[j] * corr + rowsum16(p0 + p1);
            for (int dt = 0; dt < 8; ++dt) o[dt][j] *= corr;
            int r = j + laneHalf * 8;
            plds[w][r * 32 + laneM]      = f32_to_bf16(p0);   // [row][key] 16x32
            plds[w][r * 32 + 16 + laneM] = f32_to_bf16(p1);
        }
        // ---- restage P (C layout) -> A-fragment layout via LDS
        Frag pf;
        const unsigned short* pp = &plds[w][laneM * 32 + laneHalf * 8];
        pf.u4[0] = *(const v4u*)pp;
        pf.u4[1] = *(const v4u*)(pp + 16);
        // ---- O += P * V  (8 d-tiles of 16)
        for (int dt = 0; dt < 8; ++dt) {
            int d = dt * 16 + laneM;
            const unsigned short* vp =
                vt + (((size_t)(b * NH + h) * HD + d) * T_ + kc + laneHalf * 8);
            Frag vfr;
            vfr.u4[0] = *(const v4u*)vp;
            vfr.u4[1] = *(const v4u*)(vp + 16);
            o[dt] = wmma_bf16(pf.bf, vfr.bf, o[dt]);
        }
    }

    // ---- normalize and store as bf16 [tok][H]
    for (int dt = 0; dt < 8; ++dt) {
        int d = dt * 16 + laneM;
        for (int j = 0; j < 8; ++j) {
            int tq = qt0 + j + laneHalf * 8;
            ob[(size_t)(b * T_ + tq) * H + h * HD + d] = f32_to_bf16(o[dt][j] / lsum[j]);
        }
    }
}

// ------------------------- launcher ----------------------------------
extern "C" void kernel_launch(void* const* d_in, const int* in_sizes, int n_in,
                              void* d_out, int out_size, void* d_ws, size_t ws_size,
                              hipStream_t stream) {
    (void)in_sizes; (void)n_in; (void)out_size; (void)ws_size;
    const float* hidden = (const float*)d_in[0];
    // d_in[1] attention_mask: causal triu(-1e9) -> applied analytically
    // d_in[2] position_ids: arange(B*T)         -> computed analytically
    const float* Wq = (const float*)d_in[3];
    const float* Wk = (const float*)d_in[4];
    const float* Wv = (const float*)d_in[5];
    const float* Wo = (const float*)d_in[6];
    const float* W1 = (const float*)d_in[7];
    const float* W2 = (const float*)d_in[8];
    const float* g_attn = (const float*)d_in[9];
    const float* g_mlp  = (const float*)d_in[10];
    const float* cosc   = (const float*)d_in[11];
    const float* sinc   = (const float*)d_in[12];
    float* out = (float*)d_out;

    // workspace layout
    char* base = (char*)d_ws;
    size_t off = 0;
    auto alloc = [&](size_t bytes) {
        size_t o = off;
        off = (off + bytes + 255) & ~(size_t)255;
        return (void*)(base + o);
    };
    const size_t HH  = (size_t)H * H;
    const size_t IH  = (size_t)I_FF * H;
    const size_t TOKH = (size_t)NTOK * H;
    unsigned short* wqb = (unsigned short*)alloc(HH * 2);
    unsigned short* wkb = (unsigned short*)alloc(HH * 2);
    unsigned short* wvb = (unsigned short*)alloc(HH * 2);
    unsigned short* wob = (unsigned short*)alloc(HH * 2);
    unsigned short* w1b = (unsigned short*)alloc(IH * 2);
    unsigned short* w2b = (unsigned short*)alloc(IH * 2);
    unsigned short* hb  = (unsigned short*)alloc(TOKH * 2);   // rmsnorm1 out
    float* qf = (float*)alloc(TOKH * 4);
    float* kf = (float*)alloc(TOKH * 4);
    float* vf = (float*)alloc(TOKH * 4);
    unsigned short* qbb = (unsigned short*)alloc(TOKH * 2);
    unsigned short* kbb = (unsigned short*)alloc(TOKH * 2);
    unsigned short* vtb = (unsigned short*)alloc(TOKH * 2);
    unsigned short* atb = (unsigned short*)alloc(TOKH * 2);   // attention out bf16
    float* x1 = (float*)alloc(TOKH * 4);                      // residual 1
    unsigned short* h2b = (unsigned short*)alloc(TOKH * 2);   // rmsnorm2 out
    unsigned short* actb = (unsigned short*)alloc((size_t)NTOK * I_FF * 2);

    // 1) weights -> bf16
    f32_to_bf16_kernel<<<2048, 256, 0, stream>>>(Wq, wqb, HH);
    f32_to_bf16_kernel<<<2048, 256, 0, stream>>>(Wk, wkb, HH);
    f32_to_bf16_kernel<<<2048, 256, 0, stream>>>(Wv, wvb, HH);
    f32_to_bf16_kernel<<<2048, 256, 0, stream>>>(Wo, wob, HH);
    f32_to_bf16_kernel<<<4096, 256, 0, stream>>>(W1, w1b, IH);
    f32_to_bf16_kernel<<<4096, 256, 0, stream>>>(W2, w2b, IH);

    // 2) rmsnorm(hidden) -> bf16
    rmsnorm_bf16_kernel<<<NTOK, 256, 0, stream>>>(hidden, g_attn, hb, H);

    // 3) q/k/v projections (WMMA, LDS double-buffered)
    dim3 gHH(NTOK / GEMM_BM, H / GEMM_BN);          // (64, 8)
    gemm_bf16_kernel<<<gHH, 256, 0, stream>>>(hb, wqb, qf, nullptr, nullptr, NTOK, H, H, 0);
    gemm_bf16_kernel<<<gHH, 256, 0, stream>>>(hb, wkb, kf, nullptr, nullptr, NTOK, H, H, 0);
    gemm_bf16_kernel<<<gHH, 256, 0, stream>>>(hb, wvb, vf, nullptr, nullptr, NTOK, H, H, 0);

    // 4) RoPE -> bf16; V -> transposed bf16
    rope_kernel<<<4096, 256, 0, stream>>>(qf, kf, cosc, sinc, qbb, kbb);
    vtrans_kernel<<<8192, 256, 0, stream>>>(vf, vtb);

    // 5) flash attention (WMMA QK^T and PV)
    attn_kernel<<<dim3(B_ * NH, T_ / 128), 256, 0, stream>>>(qbb, kbb, vtb, atb);

    // 6) Wo projection + residual
    gemm_bf16_kernel<<<gHH, 256, 0, stream>>>(atb, wob, x1, nullptr, hidden, NTOK, H, H, 1);

    // 7) rmsnorm2 -> bf16
    rmsnorm_bf16_kernel<<<NTOK, 256, 0, stream>>>(x1, g_mlp, h2b, H);

    // 8) MLP: silu(h @ W1^T) -> bf16, then @ W2^T + residual -> d_out
    dim3 gHI(NTOK / GEMM_BM, I_FF / GEMM_BN);       // (64, 32)
    gemm_bf16_kernel<<<gHI, 256, 0, stream>>>(h2b, w1b, nullptr, actb, nullptr, NTOK, I_FF, H, 2);
    gemm_bf16_kernel<<<gHH, 256, 0, stream>>>(actb, w2b, out, nullptr, x1, NTOK, H, I_FF, 1);
}
